// SelfAttn_75033078661890
// MI455X (gfx1250) — compile-verified, matrix operand order checked by reading, not run on hardware
//
#include <hip/hip_runtime.h>
#include <hip/hip_bf16.h>
#include <math.h>

// ---- problem constants -----------------------------------------------------
#define DD   1024
#define HH   16
#define HD   64
#define SS   2048
#define BB   2
// tokens = BB*SS = 4096

typedef __attribute__((ext_vector_type(16))) __bf16 v16bf;
typedef __attribute__((ext_vector_type(8)))  __bf16 v8bf;
typedef __attribute__((ext_vector_type(8)))  float  v8f;

// ---- helpers ---------------------------------------------------------------
__device__ __forceinline__ v8bf cvt8(const float* p) {
    float4 a = *(const float4*)p;
    float4 b = *(const float4*)(p + 4);
    v8bf r;
    r[0] = (__bf16)a.x; r[1] = (__bf16)a.y; r[2] = (__bf16)a.z; r[3] = (__bf16)a.w;
    r[4] = (__bf16)b.x; r[5] = (__bf16)b.y; r[6] = (__bf16)b.z; r[7] = (__bf16)b.w;
    return r;
}

__device__ __forceinline__ v16bf mk16(v8bf lo, v8bf hi) {
    v16bf r;
#pragma unroll
    for (int i = 0; i < 8; ++i) { r[i] = lo[i]; r[i + 8] = hi[i]; }
    return r;
}

// A/B fragment for 16x16x32 bf16 WMMA, built from one contiguous row per lane.
// rowptr: start of this lane's row; kk: 32-wide K-chunk base; kb: 0 or 8.
__device__ __forceinline__ v16bf frag_bf16(const __bf16* rowptr, int kk, int kb) {
    return mk16(*(const v8bf*)(rowptr + kk + kb),
                *(const v8bf*)(rowptr + kk + 16 + kb));
}

__device__ __forceinline__ v8f wmma_bf16(v16bf a, v16bf b, v8f c) {
    return __builtin_amdgcn_wmma_f32_16x16x32_bf16(
        /*neg_a=*/false, a, /*neg_b=*/false, b,
        /*c_mod=*/(short)0, c, /*reuse_a=*/false, /*reuse_b=*/false);
}

// reductions within each 16-lane half (xor masks < 16 never cross halves)
__device__ __forceinline__ float rmax16(float x) {
#pragma unroll
    for (int m = 1; m < 16; m <<= 1) x = fmaxf(x, __shfl_xor(x, m, 32));
    return x;
}
__device__ __forceinline__ float rsum16(float x) {
#pragma unroll
    for (int m = 1; m < 16; m <<= 1) x += __shfl_xor(x, m, 32);
    return x;
}

// ---- kernel 0: bulk f32 -> bf16 conversion (bandwidth-bound, one pass) -----
__global__ __launch_bounds__(256)
void cvt_kernel(const float* __restrict__ in, __bf16* __restrict__ out, int n8) {
    int i = blockIdx.x * blockDim.x + threadIdx.x;
    if (i < n8) *(v8bf*)(out + (size_t)i * 8) = cvt8(in + (size_t)i * 8);
}

// ---- kernel 1: QKV GEMM + bias + per-head RMS norm, bf16 in/out ------------
// grid = (3072/64, 4096/64), block = 32 (one wave). Each wave: 64x64 tile.
// q,k: (B,H,S,HD) bf16    vT: (B,H,HD,S) bf16
__global__ __launch_bounds__(32)
void qkv_kernel(const __bf16* __restrict__ x, const __bf16* __restrict__ Wqkv,
                const float* __restrict__ bqkv,
                const float* __restrict__ qn_w, const float* __restrict__ kn_w,
                __bf16* __restrict__ q, __bf16* __restrict__ k,
                __bf16* __restrict__ vT) {
    const int tid    = threadIdx.x;
    const int col    = tid & 15;
    const int kb     = (tid & 16) ? 8 : 0;
    const int rowoff = (tid & 16) ? 8 : 0;
    const int nbase  = blockIdx.x * 64;   // covers exactly one (q|k|v, head)
    const int mbase  = blockIdx.y * 64;

    v8f acc[4][4];
#pragma unroll
    for (int nt = 0; nt < 4; ++nt) {
        float bias = bqkv[nbase + nt * 16 + col];
#pragma unroll
        for (int mt = 0; mt < 4; ++mt)
#pragma unroll
            for (int j = 0; j < 8; ++j) acc[mt][nt][j] = bias;
    }

    const int m = col;
    for (int kk = 0; kk < DD; kk += 32) {
        v16bf a[4];
#pragma unroll
        for (int mt = 0; mt < 4; ++mt)
            a[mt] = frag_bf16(x + (size_t)(mbase + mt * 16 + m) * DD, kk, kb);
#pragma unroll
        for (int nt = 0; nt < 4; ++nt) {
            const __bf16* wp = Wqkv + (size_t)(nbase + nt * 16 + col) * DD;
            // speculative prefetch: OOB addresses are silently dropped, no guard
            __builtin_prefetch(wp + kk + 256, 0, 0);
            v16bf bf = frag_bf16(wp, kk, kb);
#pragma unroll
            for (int mt = 0; mt < 4; ++mt)
                acc[mt][nt] = wmma_bf16(a[mt], bf, acc[mt][nt]);
        }
    }

    const int which = nbase / DD;          // 0=q 1=k 2=v
    const int h     = (nbase % DD) / HD;

    if (which < 2) {
        const float* w = (which == 0) ? qn_w : kn_w;
        __bf16* dst    = (which == 0) ? q : k;
        float wv[4];
#pragma unroll
        for (int nt = 0; nt < 4; ++nt) wv[nt] = w[nt * 16 + col];
#pragma unroll
        for (int mt = 0; mt < 4; ++mt) {
#pragma unroll
            for (int j = 0; j < 8; ++j) {
                float ss = 0.f;
#pragma unroll
                for (int nt = 0; nt < 4; ++nt) ss += acc[mt][nt][j] * acc[mt][nt][j];
                ss = rsum16(ss);                       // full-row sum of squares
                float rinv = rsqrtf(ss * (1.0f / 64.0f) + 1e-6f);
                int token = mbase + mt * 16 + j + rowoff;
                int b = token >> 11, s = token & (SS - 1);
                size_t base = (((size_t)(b * HH + h)) * SS + s) * HD;
#pragma unroll
                for (int nt = 0; nt < 4; ++nt)
                    dst[base + nt * 16 + col] = (__bf16)(acc[mt][nt][j] * rinv * wv[nt]);
            }
        }
    } else {
#pragma unroll
        for (int mt = 0; mt < 4; ++mt) {
#pragma unroll
            for (int j = 0; j < 8; ++j) {
                int token = mbase + mt * 16 + j + rowoff;
                int b = token >> 11, s = token & (SS - 1);
#pragma unroll
                for (int nt = 0; nt < 4; ++nt) {
                    int hd = nt * 16 + col;
                    vT[(((size_t)(b * HH + h)) * HD + hd) * SS + s] = (__bf16)acc[mt][nt][j];
                }
            }
        }
    }
}

// ---- kernel 2: flash attention per (b,h,16-query tile) ---------------------
// grid = (S/16, H, B), block = 32. aout: (B,S,D) bf16 (token-major)
__global__ __launch_bounds__(32)
void attn_kernel(const __bf16* __restrict__ q, const __bf16* __restrict__ k,
                 const __bf16* __restrict__ vT, __bf16* __restrict__ aout) {
    __shared__ __attribute__((aligned(16))) __bf16 pt[16 * 32];

    const int tid    = threadIdx.x;
    const int col    = tid & 15;
    const int kb     = (tid & 16) ? 8 : 0;
    const int rowoff = (tid & 16) ? 8 : 0;
    const int st = blockIdx.x, h = blockIdx.y, b = blockIdx.z;

    const __bf16* qB = q  + (((size_t)(b * HH + h)) * SS + st * 16) * HD;
    const __bf16* kB = k  + ((size_t)(b * HH + h)) * SS * HD;
    const __bf16* vB = vT + ((size_t)(b * HH + h)) * HD * SS;

    const int m = col;
    v16bf qf0 = frag_bf16(qB + (size_t)m * HD, 0,  kb);
    v16bf qf1 = frag_bf16(qB + (size_t)m * HD, 32, kb);

    v8f o[4]; float mrow[8], lrow[8];
#pragma unroll
    for (int t = 0; t < 4; ++t)
#pragma unroll
        for (int j = 0; j < 8; ++j) o[t][j] = 0.f;
#pragma unroll
    for (int j = 0; j < 8; ++j) { mrow[j] = -INFINITY; lrow[j] = 0.f; }

    const float sc = 0.125f;   // 1/sqrt(64)

    for (int t2 = 0; t2 < SS; t2 += 32) {
        v8f s0, s1;
#pragma unroll
        for (int j = 0; j < 8; ++j) { s0[j] = 0.f; s1[j] = 0.f; }

        // scores: two 16x16 tiles over keys, K = HD = 64 in two chunks
        v16bf kf;
        kf = frag_bf16(kB + (size_t)(t2      + col) * HD, 0,  kb);
        s0 = wmma_bf16(qf0, kf, s0);
        kf = frag_bf16(kB + (size_t)(t2 + 16 + col) * HD, 0,  kb);
        s1 = wmma_bf16(qf0, kf, s1);
        kf = frag_bf16(kB + (size_t)(t2      + col) * HD, 32, kb);
        s0 = wmma_bf16(qf1, kf, s0);
        kf = frag_bf16(kB + (size_t)(t2 + 16 + col) * HD, 32, kb);
        s1 = wmma_bf16(qf1, kf, s1);

        // online softmax, row-wise (row j+rowoff owned by this 16-lane half)
#pragma unroll
        for (int j = 0; j < 8; ++j) {
            float v0 = s0[j] * sc, v1 = s1[j] * sc;
            float mx = rmax16(fmaxf(v0, v1));
            float mn = fmaxf(mrow[j], mx);
            float al = __expf(mrow[j] - mn);
            float p0 = __expf(v0 - mn), p1 = __expf(v1 - mn);
            float rs = rsum16(p0 + p1);
            lrow[j] = lrow[j] * al + rs;
            mrow[j] = mn;
#pragma unroll
            for (int t = 0; t < 4; ++t) o[t][j] *= al;
            int row = j + rowoff;
            pt[row * 32 + col]      = (__bf16)p0;
            pt[row * 32 + 16 + col] = (__bf16)p1;
        }
        __syncthreads();                       // LDS writes -> cross-lane reads

        v16bf pf = frag_bf16(&pt[m * 32], 0, kb);   // P as A-fragment (K=32)
#pragma unroll
        for (int t = 0; t < 4; ++t) {
            v16bf vf = frag_bf16(vB + (size_t)(t * 16 + col) * SS + t2, 0, kb);
            o[t] = wmma_bf16(pf, vf, o[t]);
        }
        __syncthreads();                       // protect pt before next chunk
    }

#pragma unroll
    for (int j = 0; j < 8; ++j) {
        float inv = 1.0f / lrow[j];
        int token = b * SS + st * 16 + j + rowoff;
        size_t base = (size_t)token * DD + h * HD;
#pragma unroll
        for (int t = 0; t < 4; ++t)
            aout[base + t * 16 + col] = (__bf16)(o[t][j] * inv);
    }
}

// ---- kernel 3: output projection, bf16 in, f32 out -------------------------
// grid = (1024/64, 4096/64), block = 32. out: (B,S,D) fp32
__global__ __launch_bounds__(32)
void proj_kernel(const __bf16* __restrict__ a, const __bf16* __restrict__ Wp,
                 const float* __restrict__ bp, float* __restrict__ out) {
    const int tid    = threadIdx.x;
    const int col    = tid & 15;
    const int kb     = (tid & 16) ? 8 : 0;
    const int rowoff = (tid & 16) ? 8 : 0;
    const int nbase  = blockIdx.x * 64;
    const int mbase  = blockIdx.y * 64;

    v8f acc[4][4];
#pragma unroll
    for (int nt = 0; nt < 4; ++nt) {
        float bias = bp[nbase + nt * 16 + col];
#pragma unroll
        for (int mt = 0; mt < 4; ++mt)
#pragma unroll
            for (int j = 0; j < 8; ++j) acc[mt][nt][j] = bias;
    }

    const int m = col;
    for (int kk = 0; kk < DD; kk += 32) {
        v16bf af[4];
#pragma unroll
        for (int mt = 0; mt < 4; ++mt)
            af[mt] = frag_bf16(a + (size_t)(mbase + mt * 16 + m) * DD, kk, kb);
#pragma unroll
        for (int nt = 0; nt < 4; ++nt) {
            const __bf16* wp = Wp + (size_t)(nbase + nt * 16 + col) * DD;
            __builtin_prefetch(wp + kk + 256, 0, 0);   // speculative, unguarded
            v16bf bf = frag_bf16(wp, kk, kb);
#pragma unroll
            for (int mt = 0; mt < 4; ++mt)
                acc[mt][nt] = wmma_bf16(af[mt], bf, acc[mt][nt]);
        }
    }

#pragma unroll
    for (int mt = 0; mt < 4; ++mt) {
#pragma unroll
        for (int j = 0; j < 8; ++j) {
            size_t row = (size_t)(mbase + mt * 16 + j + rowoff) * DD;
#pragma unroll
            for (int nt = 0; nt < 4; ++nt)
                out[row + nbase + nt * 16 + col] = acc[mt][nt][j];
        }
    }
}

// ---- launcher --------------------------------------------------------------
extern "C" void kernel_launch(void* const* d_in, const int* in_sizes, int n_in,
                              void* d_out, int out_size, void* d_ws, size_t ws_size,
                              hipStream_t stream) {
    const float* x     = (const float*)d_in[0];
    const float* Wqkv  = (const float*)d_in[1];
    const float* bqkv  = (const float*)d_in[2];
    const float* Wproj = (const float*)d_in[3];
    const float* bproj = (const float*)d_in[4];
    const float* qn_w  = (const float*)d_in[5];
    const float* kn_w  = (const float*)d_in[6];

    const size_t NELEM  = (size_t)BB * HH * SS * HD;  // 4,194,304 (q/k/v/aout/x)
    const size_t NWQKV  = (size_t)3 * DD * DD;        // 3,145,728
    const size_t NWPROJ = (size_t)DD * DD;            // 1,048,576

    __bf16* q     = (__bf16*)d_ws;
    __bf16* k     = q     + NELEM;
    __bf16* vT    = k     + NELEM;
    __bf16* aout  = vT    + NELEM;
    __bf16* xb    = aout  + NELEM;
    __bf16* Wqkvb = xb    + NELEM;
    __bf16* Wprjb = Wqkvb + NWQKV;                    // total 48 MB

    // one-pass f32 -> bf16 of activations + weights (bandwidth-bound)
    cvt_kernel<<<(int)(NELEM  / 8 / 256), 256, 0, stream>>>(x,     xb,    (int)(NELEM  / 8));
    cvt_kernel<<<(int)(NWQKV  / 8 / 256), 256, 0, stream>>>(Wqkv,  Wqkvb, (int)(NWQKV  / 8));
    cvt_kernel<<<(int)(NWPROJ / 8 / 256), 256, 0, stream>>>(Wproj, Wprjb, (int)(NWPROJ / 8));

    qkv_kernel<<<dim3(48, 64), 32, 0, stream>>>(xb, Wqkvb, bqkv, qn_w, kn_w, q, k, vT);
    attn_kernel<<<dim3(SS / 16, HH, BB), 32, 0, stream>>>(q, k, vT, aout);
    proj_kernel<<<dim3(16, 64), 32, 0, stream>>>(aout, Wprjb, bproj, (float*)d_out);
}
